// DepthlessTransformer_66769561584209
// MI455X (gfx1250) — compile-verified, matrix operand order checked by reading, not run on hardware
//
#include <hip/hip_runtime.h>
#include <math.h>

// ---------------------------------------------------------------------------
// DepthlessTransformer forward for MI455X (gfx1250), wave32 + WMMA bf16.
// Big GEMMs: 32x64 register-blocked tiles per wave (8 v_wmma per K-step).
// GEMM operands pre-converted to bf16 (weights once per launch, activations
// at their producers) -> inner loop is pure b128 loads + WMMA, no cvt.
// Workspace requirement: ~326 MB.
// ---------------------------------------------------------------------------

#define DD   512
#define HH   8
#define DH   64
#define DI   512
#define LL   6
#define NEXC 6
#define FFI  1365
#define FFIP 1376          // FFI padded to a multiple of 32 for WMMA K loop
#define VV   32000
#define BB   2
#define NN   192
#define TOK  (LL * BB * NN)   // 2304 token-rows (l,b,n)
#define TOKB (BB * NN)        // 384
#define NMSG 13               // 1 + 2*NEXC messages
#define MTFIN (NMSG * LL)     // 78
#define EPSF 1.1920929e-07f

typedef __attribute__((ext_vector_type(16))) __bf16 v16bf;
typedef __attribute__((ext_vector_type(8)))  __bf16 v8bf;
typedef __attribute__((ext_vector_type(8)))  float  v8f;

// ---- WMMA fragment helpers (16x16x32 bf16, wave32) -------------------------
// A-matrix 16x32: lanes 0-15 hold row M=lane, K in {0..7,16..23};
// lanes 16-31 hold row M=lane-16, K in {8..15,24..31}. B symmetric from
// row-major [Nc,K] weights (B[k][n] = W[n][k]).

// bf16 fragment: two 16B loads (global_load_b128), no conversion.
__device__ __forceinline__ v16bf frag_load_bf(const __bf16* p) {
  const int lane = threadIdx.x & 31;
  const int kb = (lane < 16) ? 0 : 8;
  const v8bf lo = *(const v8bf*)(p + kb);
  const v8bf hi = *(const v8bf*)(p + 16 + kb);
  v16bf f;
#pragma unroll
  for (int j = 0; j < 8; ++j) { f[j] = lo[j]; f[j + 8] = hi[j]; }
  return f;
}

// fp32 contiguous fragment: 4x float4 + pack-convert.
__device__ __forceinline__ v16bf frag_load4(const float* p) {
  const int lane = threadIdx.x & 31;
  const int kb = (lane < 16) ? 0 : 8;
  const float4 x0 = *(const float4*)(p + kb);
  const float4 x1 = *(const float4*)(p + kb + 4);
  const float4 x2 = *(const float4*)(p + kb + 16);
  const float4 x3 = *(const float4*)(p + kb + 20);
  v16bf f;
  f[0]  = (__bf16)x0.x; f[1]  = (__bf16)x0.y; f[2]  = (__bf16)x0.z; f[3]  = (__bf16)x0.w;
  f[4]  = (__bf16)x1.x; f[5]  = (__bf16)x1.y; f[6]  = (__bf16)x1.z; f[7]  = (__bf16)x1.w;
  f[8]  = (__bf16)x2.x; f[9]  = (__bf16)x2.y; f[10] = (__bf16)x2.z; f[11] = (__bf16)x2.w;
  f[12] = (__bf16)x3.x; f[13] = (__bf16)x3.y; f[14] = (__bf16)x3.z; f[15] = (__bf16)x3.w;
  return f;
}

// Strided fp32 fragment (scalar loads) for kStride != 1 operands.
__device__ __forceinline__ v16bf frag_load_s(const float* p, long long ks) {
  const int lane = threadIdx.x & 31;
  const int kb = (lane < 16) ? 0 : 8;
  v16bf f;
#pragma unroll
  for (int j = 0; j < 8; ++j) {
    f[j]     = (__bf16)p[(long long)(kb + j) * ks];
    f[j + 8] = (__bf16)p[(long long)(16 + kb + j) * ks];
  }
  return f;
}

// fp32 fragment with fused RMSNorm: elem * rstd(row) * norm_w[k].
__device__ __forceinline__ v16bf frag_load_norm(const float* p, const float* nw, float rs) {
  const int lane = threadIdx.x & 31;
  const int kb = (lane < 16) ? 0 : 8;
  const float4 x0 = *(const float4*)(p + kb);
  const float4 x1 = *(const float4*)(p + kb + 4);
  const float4 x2 = *(const float4*)(p + kb + 16);
  const float4 x3 = *(const float4*)(p + kb + 20);
  const float4 w0 = *(const float4*)(nw + kb);
  const float4 w1 = *(const float4*)(nw + kb + 4);
  const float4 w2 = *(const float4*)(nw + kb + 16);
  const float4 w3 = *(const float4*)(nw + kb + 20);
  v16bf f;
  f[0]  = (__bf16)(x0.x * rs * w0.x); f[1]  = (__bf16)(x0.y * rs * w0.y);
  f[2]  = (__bf16)(x0.z * rs * w0.z); f[3]  = (__bf16)(x0.w * rs * w0.w);
  f[4]  = (__bf16)(x1.x * rs * w1.x); f[5]  = (__bf16)(x1.y * rs * w1.y);
  f[6]  = (__bf16)(x1.z * rs * w1.z); f[7]  = (__bf16)(x1.w * rs * w1.w);
  f[8]  = (__bf16)(x2.x * rs * w2.x); f[9]  = (__bf16)(x2.y * rs * w2.y);
  f[10] = (__bf16)(x2.z * rs * w2.z); f[11] = (__bf16)(x2.w * rs * w2.w);
  f[12] = (__bf16)(x3.x * rs * w3.x); f[13] = (__bf16)(x3.y * rs * w3.y);
  f[14] = (__bf16)(x3.z * rs * w3.z); f[15] = (__bf16)(x3.w * rs * w3.w);
  return f;
}

__device__ __forceinline__ v8f wmma_bf16(v16bf a, v16bf b, v8f c) {
  return __builtin_amdgcn_wmma_f32_16x16x32_bf16(false, a, false, b, (short)0, c,
                                                 false, false);
}

__device__ __forceinline__ float wave_sum(float s) {
#pragma unroll
  for (int off = 16; off > 0; off >>= 1) s += __shfl_xor(s, off, 32);
  return s;
}
__device__ __forceinline__ float wave_max(float s) {
#pragma unroll
  for (int off = 16; off > 0; off >>= 1) s = fmaxf(s, __shfl_xor(s, off, 32));
  return s;
}

// ---------------------------------------------------------------------------
// Generic GEMM: C[M,Nc] = A[M,K] * W[Nc,K]^T (+ bias). A and W are bf16,
// C/bias fp32. One wave per 32x64 tile: 2 A-frags x 4 B-frags -> 8 WMMAs
// per 32-wide K step, 12x b128 loads, zero conversions.
// K % 32 == 0; M % 32 == 0; Nc may be ragged; rows 16B-aligned.
// ---------------------------------------------------------------------------
__global__ __launch_bounds__(32) void gemm_wmma(
    float* __restrict__ C, const __bf16* __restrict__ A,
    const __bf16* __restrict__ W, const float* __restrict__ bias,
    int M, int Nc, int K, int lda, int ldw, int ldc,
    int m_per_block, long long wBlockStride, int biasBlockStride) {
  const int lane = threadIdx.x & 31;
  const int tN = blockIdx.x, tM = blockIdx.y;
  const int blk = (tM * 32) / m_per_block;
  const __bf16* Wp = W + (long long)blk * wBlockStride;

  const __bf16* ap0 = A + (long long)(tM * 32 + (lane & 15)) * lda;
  const __bf16* ap1 = ap0 + (long long)16 * lda;
  const __bf16* bp[4];
#pragma unroll
  for (int t = 0; t < 4; ++t) {
    int brow = tN * 64 + t * 16 + (lane & 15);
    if (brow >= Nc) brow = Nc - 1;
    bp[t] = Wp + (long long)brow * ldw;
  }

  v8f acc[2][4] = {};
  for (int k0 = 0; k0 < K; k0 += 32) {
    const v16bf a0 = frag_load_bf(ap0 + k0);
    const v16bf a1 = frag_load_bf(ap1 + k0);
#pragma unroll
    for (int t = 0; t < 4; ++t) {
      const v16bf b = frag_load_bf(bp[t] + k0);
      acc[0][t] = wmma_bf16(a0, b, acc[0][t]);
      acc[1][t] = wmma_bf16(a1, b, acc[1][t]);
    }
  }

#pragma unroll
  for (int mi = 0; mi < 2; ++mi) {
    const int m0 = tM * 32 + mi * 16 + ((lane < 16) ? 0 : 8);
#pragma unroll
    for (int t = 0; t < 4; ++t) {
      const int n = tN * 64 + t * 16 + (lane & 15);
      if (n < Nc) {
        const float bv = bias ? bias[(long long)blk * biasBlockStride + n] : 0.0f;
#pragma unroll
        for (int r = 0; r < 8; ++r)
          C[(long long)(m0 + r) * ldc + n] = acc[mi][t][r] + bv;
      }
    }
  }
}

// ---------------------------------------------------------------------------
// Blocked self-attention scores: sim[z][i][j] = sum_d q[i,d]*k[j,d], d<64.
// z = (l*B+b)*H + h ; grid (12,12,96), one wave per 16x16 tile.
// ---------------------------------------------------------------------------
__global__ __launch_bounds__(32) void attn_scores(
    float* __restrict__ sim, const float* __restrict__ q,
    const float* __restrict__ kv) {
  const int lane = threadIdx.x & 31;
  const int z = blockIdx.z, lb = z >> 3, h = z & 7;
  const int tN = blockIdx.x, tM = blockIdx.y;
  const float* Q  = q  + (long long)lb * NN * DI + h * DH;
  const float* Kp = kv + (long long)lb * NN * (2 * DI) + h * DH;
  const float* ap = Q  + (long long)(tM * 16 + (lane & 15)) * DI;
  const float* bp = Kp + (long long)(tN * 16 + (lane & 15)) * (2 * DI);
  v8f acc = {};
#pragma unroll
  for (int k0 = 0; k0 < DH; k0 += 32) {
    acc = wmma_bf16(frag_load4(ap + k0), frag_load4(bp + k0), acc);
  }
  float* Cp = sim + (long long)z * NN * NN;
  const int n = tN * 16 + (lane & 15);
  const int m0 = tM * 16 + ((lane < 16) ? 0 : 8);
#pragma unroll
  for (int r = 0; r < 8; ++r) Cp[(long long)(m0 + r) * NN + n] = acc[r];
}

// o[i, h*64+d] = sum_j attn[i,j] * v[j, h*64+d]; strided B (kStride=1024).
// Output stored bf16 (consumed only as GEMM A operand).
__global__ __launch_bounds__(32) void attn_av(
    __bf16* __restrict__ o, const float* __restrict__ attn,
    const float* __restrict__ kv) {
  const int lane = threadIdx.x & 31;
  const int z = blockIdx.z, lb = z >> 3, h = z & 7;
  const int tN = blockIdx.x, tM = blockIdx.y;
  const float* Ap = attn + (long long)z * NN * NN +
                    (long long)(tM * 16 + (lane & 15)) * NN;
  const float* Vb = kv + (long long)lb * NN * (2 * DI) + DI + h * DH;
  const float* bp = Vb + (tN * 16 + (lane & 15));  // row = d, stride over j
  v8f acc = {};
  for (int k0 = 0; k0 < NN; k0 += 32) {
    acc = wmma_bf16(frag_load4(Ap + k0),
                    frag_load_s(bp + (long long)k0 * (2 * DI), 2 * DI), acc);
  }
  const int n = tN * 16 + (lane & 15);
  const int m0 = tM * 16 + ((lane < 16) ? 0 : 8);
#pragma unroll
  for (int r = 0; r < 8; ++r)
    o[(long long)(lb * NN + m0 + r) * DI + h * DH + n] = (__bf16)acc[r];
}

// ---------------------------------------------------------------------------
// Pooled-context KV projection: kvb[(b*N+n)*Mt + mi][f] =
//   sum_d rmsnorm(msgs[mi, b, n, :], res_norm)[d] * res_kv_w[f][d]
// mi = packed message index (m*L + l); rstd precomputed per message row.
// 32x64 register-blocked; fp32 A with fused norm, bf16 B.
// ---------------------------------------------------------------------------
__global__ __launch_bounds__(32) void pooled_kv(
    float* __restrict__ kvb, const float* __restrict__ msgs,
    const float* __restrict__ rstd, const float* __restrict__ nw,
    const __bf16* __restrict__ Wkv, int Mt) {
  const int lane = threadIdx.x & 31;
  const int tN = blockIdx.x, tM = blockIdx.y;

  const float* ap[2];
  float rs[2];
#pragma unroll
  for (int mi = 0; mi < 2; ++mi) {
    const int r = tM * 32 + mi * 16 + (lane & 15);
    const int m = r % Mt;
    const int bn = r / Mt;
    const long long mrow = ((long long)m * BB + bn / NN) * NN + (bn % NN);
    ap[mi] = msgs + mrow * DD;
    rs[mi] = rstd[mrow];
  }
  const __bf16* bp[4];
#pragma unroll
  for (int t = 0; t < 4; ++t)
    bp[t] = Wkv + (long long)(tN * 64 + t * 16 + (lane & 15)) * DD;

  v8f acc[2][4] = {};
  for (int k0 = 0; k0 < DD; k0 += 32) {
    const v16bf a0 = frag_load_norm(ap[0] + k0, nw + k0, rs[0]);
    const v16bf a1 = frag_load_norm(ap[1] + k0, nw + k0, rs[1]);
#pragma unroll
    for (int t = 0; t < 4; ++t) {
      const v16bf b = frag_load_bf(bp[t] + k0);
      acc[0][t] = wmma_bf16(a0, b, acc[0][t]);
      acc[1][t] = wmma_bf16(a1, b, acc[1][t]);
    }
  }
#pragma unroll
  for (int mi = 0; mi < 2; ++mi) {
    const int m0 = tM * 32 + mi * 16 + ((lane < 16) ? 0 : 8);
#pragma unroll
    for (int t = 0; t < 4; ++t) {
      const int n = tN * 64 + t * 16 + (lane & 15);
#pragma unroll
      for (int r = 0; r < 8; ++r)
        kvb[(long long)(m0 + r) * (2 * DI) + n] = acc[mi][t][r];
    }
  }
}

// ---------------------------------------------------------------------------
// Small VALU kernels
// ---------------------------------------------------------------------------
// RMSNorm, bf16 output (consumed only as GEMM A operand).
__global__ __launch_bounds__(32) void rmsnorm_rows(
    __bf16* __restrict__ out, const float* __restrict__ x,
    const float* __restrict__ w, int in_stride, int rows_per_block) {
  const int lane = threadIdx.x;
  const long long r = blockIdx.x;
  const float* xr = x + r * (long long)in_stride;
  const float* wr = w + (long long)(r / rows_per_block) * DD;
  float ss = 0.f;
#pragma unroll
  for (int j = 0; j < 16; ++j) { float t = xr[lane + 32 * j]; ss += t * t; }
  ss = wave_sum(ss);
  const float rs = rsqrtf(ss * (1.0f / DD) + EPSF);
  __bf16* orow = out + r * (long long)DD;
#pragma unroll
  for (int j = 0; j < 16; ++j)
    orow[lane + 32 * j] = (__bf16)(xr[lane + 32 * j] * rs * wr[lane + 32 * j]);
}

__global__ __launch_bounds__(32) void row_rstd(
    float* __restrict__ rstd, const float* __restrict__ x) {
  const int lane = threadIdx.x;
  const float* xr = x + (long long)blockIdx.x * DD;
  float ss = 0.f;
#pragma unroll
  for (int j = 0; j < 16; ++j) { float t = xr[lane + 32 * j]; ss += t * t; }
  ss = wave_sum(ss);
  if (lane == 0) rstd[blockIdx.x] = rsqrtf(ss * (1.0f / DD) + EPSF);
}

// Per-head key RMSNorm over DH=64, in place on kvb's K half.
__global__ __launch_bounds__(32) void key_norm(
    float* __restrict__ kvb, const float* __restrict__ knw) {
  const int lane = threadIdx.x;
  const long long tm = blockIdx.x >> 3;
  const int h = blockIdx.x & 7;
  float* p = kvb + tm * (2 * DI) + h * DH;
  const float a = p[lane], b = p[32 + lane];
  float ss = wave_sum(a * a + b * b);
  const float rs = rsqrtf(ss * (1.0f / DH) + EPSF);
  p[lane] = a * rs * knw[lane];
  p[32 + lane] = b * rs * knw[32 + lane];
}

__global__ __launch_bounds__(32) void softmax192(float* __restrict__ x) {
  const int lane = threadIdx.x;
  float* row = x + (long long)blockIdx.x * NN;
  float v[6];
  float mx = -3.4e38f;
#pragma unroll
  for (int j = 0; j < 6; ++j) { v[j] = row[lane + 32 * j]; mx = fmaxf(mx, v[j]); }
  mx = wave_max(mx);
  float s = 0.f;
#pragma unroll
  for (int j = 0; j < 6; ++j) { v[j] = __expf(v[j] - mx); s += v[j]; }
  s = wave_sum(s);
  const float inv = 1.0f / s;
#pragma unroll
  for (int j = 0; j < 6; ++j) row[lane + 32 * j] = v[j] * inv;
}

// Single-query pooled attention over Mt messages (unscaled, softmax over m).
// mode 0: r=(l,b,n) query rows, ctx_b=((l*B+b)/L)  [faithful reference quirk]
// mode 1: r=(b,n) readout rows, ctx_b=b
// Output stored bf16 (consumed only as GEMM A operand).
__global__ __launch_bounds__(32) void pooled_attn(
    __bf16* __restrict__ o, const float* __restrict__ qp,
    const float* __restrict__ kvb, int Mt, int mode) {
  const int lane = threadIdx.x;
  const int r = blockIdx.x;
  const int n = r % NN;
  const int row = r / NN;
  const int ctx_b = mode ? row : (row / LL);
  const float* q = qp + (long long)r * DI;
  const float* kvtok = kvb + ((long long)(ctx_b * NN + n) * Mt) * (2 * DI);
  float sim[80];
  for (int h = 0; h < HH; ++h) {
    const float* qh = q + h * DH;
    const float qa = qh[lane], qb = qh[32 + lane];
    for (int m = 0; m < Mt; ++m) {
      const float* kp = kvtok + (long long)m * (2 * DI) + h * DH;
      sim[m] = wave_sum(qa * kp[lane] + qb * kp[32 + lane]);
    }
    float mx = -3.4e38f;
    for (int m = 0; m < Mt; ++m) mx = fmaxf(mx, sim[m]);
    float den = 0.f;
    for (int m = 0; m < Mt; ++m) { sim[m] = __expf(sim[m] - mx); den += sim[m]; }
    const float inv = 1.0f / den;
    float oa = 0.f, ob = 0.f;
    for (int m = 0; m < Mt; ++m) {
      const float* vp = kvtok + (long long)m * (2 * DI) + DI + h * DH;
      const float p = sim[m] * inv;
      oa += p * vp[lane];
      ob += p * vp[32 + lane];
    }
    o[(long long)r * DI + h * DH + lane] = (__bf16)oa;
    o[(long long)r * DI + h * DH + 32 + lane] = (__bf16)ob;
  }
}

__global__ void bcast_tokens(float* __restrict__ toks, float* __restrict__ msg0,
                             const float* __restrict__ tokens) {
  const long long i = (long long)blockIdx.x * blockDim.x + threadIdx.x;
  if (i >= (long long)TOK * DD) return;
  const float v = tokens[i % ((long long)BB * NN * DD)];
  toks[i] = v;
  msg0[i] = v;
}

// GEGLU: act[r,f] = z[r,f] * gelu_exact(z[r,FFI+f]); pad cols [FFI,FFIP)=0.
// bf16 output.
__global__ void geglu(__bf16* __restrict__ act, const float* __restrict__ z) {
  const long long i = (long long)blockIdx.x * blockDim.x + threadIdx.x;
  if (i >= (long long)TOK * FFIP) return;
  const int r = (int)(i / FFIP), f = (int)(i % FFIP);
  float v = 0.f;
  if (f < FFI) {
    const float s = z[(long long)r * (2 * FFI) + f];
    const float g = z[(long long)r * (2 * FFI) + FFI + f];
    v = s * (0.5f * g * (1.0f + erff(g * 0.70710678118654752f)));
  }
  act[i] = (__bf16)v;
}

// Repack ff_vals_w [L*D, FFI] into zero-padded bf16 [L*D, FFIP].
__global__ void pad_ffvals(__bf16* __restrict__ outw, const float* __restrict__ w) {
  const long long i = (long long)blockIdx.x * blockDim.x + threadIdx.x;
  if (i >= (long long)LL * DD * FFIP) return;
  const int r = (int)(i / FFIP), f = (int)(i % FFIP);
  outw[i] = (__bf16)((f < FFI) ? w[(long long)r * FFI + f] : 0.f);
}

// fp32 -> bf16 elementwise weight conversion.
__global__ void cvt_bf16(__bf16* __restrict__ dst, const float* __restrict__ src,
                         long long n) {
  const long long i = (long long)blockIdx.x * blockDim.x + threadIdx.x;
  if (i < n) dst[i] = (__bf16)src[i];
}

// ---------------------------------------------------------------------------
// Host orchestration
// ---------------------------------------------------------------------------
static inline void gemm(hipStream_t s, float* C, const __bf16* A, const __bf16* W,
                        const float* bias, int M, int Nc, int K, int lda,
                        int ldw, int ldc, int mpb, long long wBlk, int bBlk) {
  dim3 g((Nc + 63) / 64, M / 32, 1);
  gemm_wmma<<<g, 32, 0, s>>>(C, A, W, bias, M, Nc, K, lda, ldw, ldc, mpb, wBlk, bBlk);
}

static inline void cvt(hipStream_t s, __bf16* dst, const float* src, long long n) {
  cvt_bf16<<<(unsigned)((n + 255) / 256), 256, 0, s>>>(dst, src, n);
}

extern "C" void kernel_launch(void* const* d_in, const int* in_sizes, int n_in,
                              void* d_out, int out_size, void* d_ws, size_t ws_size,
                              hipStream_t stream) {
  (void)in_sizes; (void)n_in; (void)out_size; (void)ws_size;
  const float* tokens   = (const float*)d_in[0];
  const float* attn_nw  = (const float*)d_in[1];
  const float* attn_qw  = (const float*)d_in[2];
  const float* attn_kvw = (const float*)d_in[3];
  const float* attn_ow  = (const float*)d_in[4];
  const float* ff_nw    = (const float*)d_in[5];
  const float* ff_kw    = (const float*)d_in[6];
  const float* ff_kb    = (const float*)d_in[7];
  const float* ff_vw    = (const float*)d_in[8];
  const float* ff_vb    = (const float*)d_in[9];
  const float* res_nw   = (const float*)d_in[10];
  const float* res_knw  = (const float*)d_in[11];
  const float* res_qw   = (const float*)d_in[12];
  const float* res_kvw  = (const float*)d_in[13];
  const float* res_ow   = (const float*)d_in[14];
  const float* q_read   = (const float*)d_in[15];
  const float* ro_nw    = (const float*)d_in[16];
  const float* ro_w     = (const float*)d_in[17];
  float* out = (float*)d_out;

  // Workspace layout (float units): ~326 MB.
  float* ws = (float*)d_ws;
  const long long SZ = (long long)TOK * DD;
  long long off = 0;
  float* msgs  = ws + off; off += (long long)NMSG * SZ;     // 13 message tensors
  float* rstdm = ws + off; off += (long long)NMSG * TOK;    // per-msg-row rstd
  float* toks  = ws + off; off += SZ;
  float* qbuf  = ws + off; off += SZ;
  float* kva   = ws + off; off += 2 * SZ;
  float* sim   = ws + off; off += (long long)LL * BB * HH * NN * NN;
  float* ffz   = ws + off; off += (long long)TOK * (2 * FFI);
  float* kvb   = ws + off; off += (long long)TOKB * MTFIN * (2 * DI);
  // bf16 regions (sizes in bf16 elements; advance off by half, wffk by half + 1)
  __bf16* hbuf  = (__bf16*)(ws + off); off += SZ / 2;
  __bf16* obuf  = (__bf16*)(ws + off); off += SZ / 2;
  __bf16* ffact = (__bf16*)(ws + off); off += ((long long)TOK * FFIP) / 2;
  __bf16* wqa   = (__bf16*)(ws + off); off += ((long long)LL * DI * DD) / 2;
  __bf16* wkva  = (__bf16*)(ws + off); off += ((long long)LL * 2 * DI * DD) / 2;
  __bf16* wow   = (__bf16*)(ws + off); off += ((long long)LL * DD * DI) / 2;
  __bf16* wffk  = (__bf16*)(ws + off); off += ((long long)LL * 2 * FFI * DD + 2) / 2;
  __bf16* wffv  = (__bf16*)(ws + off); off += ((long long)LL * DD * FFIP) / 2;
  __bf16* wrq   = (__bf16*)(ws + off); off += ((long long)DI * DD) / 2;
  __bf16* wrkv  = (__bf16*)(ws + off); off += ((long long)2 * DI * DD) / 2;
  __bf16* wrow  = (__bf16*)(ws + off); off += ((long long)DD * DI) / 2;
  __bf16* wro   = (__bf16*)(ws + off); off += ((long long)VV * DD) / 2;

  // ---- setup: broadcast tokens, pre-convert all weights to bf16 ----
  {
    long long n = (long long)TOK * DD;
    bcast_tokens<<<(unsigned)((n + 255) / 256), 256, 0, stream>>>(toks, msgs, tokens);
    row_rstd<<<TOK, 32, 0, stream>>>(rstdm, msgs);
    long long np = (long long)LL * DD * FFIP;
    pad_ffvals<<<(unsigned)((np + 255) / 256), 256, 0, stream>>>(wffv, ff_vw);
    cvt(stream, wqa,  attn_qw,  (long long)LL * DI * DD);
    cvt(stream, wkva, attn_kvw, (long long)LL * 2 * DI * DD);
    cvt(stream, wow,  attn_ow,  (long long)LL * DD * DI);
    cvt(stream, wffk, ff_kw,    (long long)LL * 2 * FFI * DD);
    cvt(stream, wrq,  res_qw,   (long long)DI * DD);
    cvt(stream, wrkv, res_kvw,  (long long)2 * DI * DD);
    cvt(stream, wrow, res_ow,   (long long)DD * DI);
    cvt(stream, wro,  ro_w,     (long long)VV * DD);
  }

  for (int i = 0; i < NEXC; ++i) {
    // ---- blocked self-attention -> messages[1+2i] ----
    rmsnorm_rows<<<TOK, 32, 0, stream>>>(hbuf, toks, attn_nw, DD, TOKB);
    gemm(stream, qbuf, hbuf, wqa, nullptr, TOK, DI, DD, DD, DD, DI,
         TOKB, (long long)DI * DD, 0);
    gemm(stream, kva, hbuf, wkva, nullptr, TOK, 2 * DI, DD, DD, DD, 2 * DI,
         TOKB, (long long)2 * DI * DD, 0);
    attn_scores<<<dim3(NN / 16, NN / 16, LL * BB * HH), 32, 0, stream>>>(sim, qbuf, kva);
    softmax192<<<LL * BB * HH * NN, 32, 0, stream>>>(sim);
    attn_av<<<dim3(DH / 16, NN / 16, LL * BB * HH), 32, 0, stream>>>(obuf, sim, kva);
    float* att_msg = msgs + (long long)(1 + 2 * i) * SZ;
    gemm(stream, att_msg, obuf, wow, nullptr, TOK, DD, DI, DI, DI, DD,
         TOKB, (long long)DD * DI, 0);
    row_rstd<<<TOK, 32, 0, stream>>>(rstdm + (long long)(1 + 2 * i) * TOK, att_msg);

    // ---- GEGLU feed-forward -> messages[2+2i] ----
    rmsnorm_rows<<<TOK, 32, 0, stream>>>(hbuf, toks, ff_nw, DD, TOKB);
    gemm(stream, ffz, hbuf, wffk, ff_kb, TOK, 2 * FFI, DD, DD, DD, 2 * FFI,
         TOKB, (long long)2 * FFI * DD, 2 * FFI);
    {
      long long n = (long long)TOK * FFIP;
      geglu<<<(unsigned)((n + 255) / 256), 256, 0, stream>>>(ffact, ffz);
    }
    float* ff_msg = msgs + (long long)(2 + 2 * i) * SZ;
    gemm(stream, ff_msg, ffact, wffv, ff_vb, TOK, DD, FFIP, FFIP, FFIP, DD,
         TOKB, (long long)DD * FFIP, DD);
    row_rstd<<<TOK, 32, 0, stream>>>(rstdm + (long long)(2 + 2 * i) * TOK, ff_msg);

    // ---- pooled message-exchange attention (skip on last) ----
    if (i < NEXC - 1) {
      const int Mt = (3 + 2 * i) * LL;  // packed message count: 18,30,42,54,66
      rmsnorm_rows<<<TOK, 32, 0, stream>>>(hbuf, toks, res_nw, DD, TOK);
      gemm(stream, qbuf, hbuf, wrq, nullptr, TOK, DI, DD, DD, DD, DI,
           TOK, 0, 0);
      pooled_kv<<<dim3((2 * DI) / 64, (TOKB * Mt) / 32), 32, 0, stream>>>(
          kvb, msgs, rstdm, res_nw, wrkv, Mt);
      key_norm<<<TOKB * Mt * HH, 32, 0, stream>>>(kvb, res_knw);
      pooled_attn<<<TOK, 32, 0, stream>>>(obuf, qbuf, kvb, Mt, 0);
      gemm(stream, toks, obuf, wrow, nullptr, TOK, DD, DI, DI, DI, DD,
           TOK, 0, 0);
    }
  }

  // ---- readout: query_readout attends over all 78 messages ----
  rmsnorm_rows<<<TOKB, 32, 0, stream>>>(hbuf, q_read, res_nw, 0, TOKB);
  gemm(stream, qbuf, hbuf, wrq, nullptr, TOKB, DI, DD, DD, DD, DI, TOKB, 0, 0);
  pooled_kv<<<dim3((2 * DI) / 64, (TOKB * MTFIN) / 32), 32, 0, stream>>>(
      kvb, msgs, rstdm, res_nw, wrkv, MTFIN);
  key_norm<<<TOKB * MTFIN * HH, 32, 0, stream>>>(kvb, res_knw);
  pooled_attn<<<TOKB, 32, 0, stream>>>(obuf, qbuf, kvb, MTFIN, 1);
  gemm(stream, toks, obuf, wrow, nullptr, TOKB, DD, DI, DI, DI, DD, TOKB, 0, 0);

  // ---- logits: rmsnorm -> [384, 32000] GEMM ----
  rmsnorm_rows<<<TOKB, 32, 0, stream>>>(hbuf, toks, ro_nw, DD, TOKB);
  gemm(stream, out, hbuf, wro, nullptr, TOKB, VV, DD, DD, DD, VV, TOKB, 0, 0);
}
